// EnhancedGNN_72868415144304
// MI455X (gfx1250) — matrix operand order repaired and checked by reading, not compiled, not run on hardware
//
#include <hip/hip_runtime.h>
#include <hip/hip_bf16.h>
#include <math.h>

// ---------------- problem constants ----------------
#define N_NODES 50000
#define N_EDGES 800000
#define IN_DIM  256
#define HID     32
#define HEADS   4
#define HO      128
#define OUT_DIM 128
#define EDGE_DIM 32
#define L_LAYERS 3
#define NG      32

typedef __attribute__((ext_vector_type(16))) _Float16     v16h;
typedef __attribute__((ext_vector_type(8)))  float        v8f;
typedef __attribute__((ext_vector_type(4)))  unsigned int u32x4;
typedef __attribute__((ext_vector_type(4)))  float        f32x4;

union Frag {
  v16h  h;
  u32x4 q[2];
};

// ---------------- WMMA wrapper ----------------
__device__ inline v8f wmma_f32_16x16x32(v16h a, v16h b, v8f c) {
  return __builtin_amdgcn_wmma_f32_16x16x32_f16(false, a, false, b, (short)0, c,
                                                false, false);
}

// ---------------- weight pre-pack: f32 [K,N] -> per-lane fragment order -------
// B-fragment layout (32x16 KxN, f16): lane n=lane&15, kb=(lane>>4)*16,
// dword r holds {K=kb+2r (lo), K=kb+2r+1 (hi)}.  8 dwords contiguous per lane.
__global__ void __launch_bounds__(32)
pack_b_frags(const float* __restrict__ W, unsigned int* __restrict__ out,
             int K, int N) {
  int tile   = blockIdx.x;              // tile = ks * ntiles + nt
  int ntiles = N >> 4;
  int ks     = tile / ntiles;
  int nt     = tile % ntiles;
  int lane   = threadIdx.x & 31;
  int n      = nt * 16 + (lane & 15);
  int kb     = ks * 32 + ((lane >> 4) * 16);
  unsigned int* o = out + ((size_t)tile * 32 + lane) * 8;
#pragma unroll
  for (int r = 0; r < 8; ++r) {
    union { _Float16 h[2]; unsigned int u; } cv;
    cv.h[0] = (_Float16)W[(size_t)(kb + 2 * r) * N + n];
    cv.h[1] = (_Float16)W[(size_t)(kb + 2 * r + 1) * N + n];
    o[r] = cv.u;
  }
}

// ---------------- f32 -> f16 stream conversion --------------------------------
__global__ void cvt_to_f16(const float* __restrict__ in, _Float16* __restrict__ out,
                           long n) {
  long i = blockIdx.x * (long)blockDim.x + threadIdx.x;
  if (i < n) out[i] = (_Float16)in[i];
}

// ---------------- A-fragment loaders -----------------------------------------
// f16 A [M,K]: lane m=lane&15; K=kb..kb+15 contiguous (kb=(lane>>4)*16).
__device__ inline v16h load_a_f16(const _Float16* __restrict__ A, int K,
                                  int row0, int ks, int lane) {
  int m   = lane & 15;
  int kb4 = lane >> 4;  // u32x4 index: 0 or 1
  const u32x4* p = (const u32x4*)(A + (size_t)(row0 + m) * K + ks * 32);
  Frag f;
  f.q[0] = p[2 * kb4];        // K = kb .. kb+7
  f.q[1] = p[2 * kb4 + 1];    // K = kb+8 .. kb+15
  return f.h;
}

// f32 A [M,K]: 4 x b128 loads + pack to f16.
__device__ inline v16h load_a_f32(const float* __restrict__ A, int K,
                                  int row0, int ks, int lane) {
  int m  = lane & 15;
  int kb = (lane >> 4) * 16;
  const f32x4* p = (const f32x4*)(A + (size_t)(row0 + m) * K + ks * 32 + kb);
  f32x4 lo0 = p[0];   // K = kb   .. kb+3
  f32x4 lo1 = p[1];   // K = kb+4 .. kb+7
  f32x4 hi0 = p[2];   // K = kb+8 .. kb+11
  f32x4 hi1 = p[3];   // K = kb+12.. kb+15
  v16h a;
#pragma unroll
  for (int t = 0; t < 4; ++t) {
    a[t]      = (_Float16)lo0[t];
    a[4 + t]  = (_Float16)lo1[t];
    a[8 + t]  = (_Float16)hi0[t];
    a[12 + t] = (_Float16)hi1[t];
  }
  return a;
}

__device__ inline v16h load_b_frag(const unsigned int* __restrict__ Bf,
                                   int ntiles, int ks, int nt, int lane) {
  const u32x4* bp =
      (const u32x4*)(Bf + ((size_t)(ks * ntiles + nt) * 32 + lane) * 8);
  Frag f;
  f.q[0] = bp[0];
  f.q[1] = bp[1];
  return f.h;
}

// ---------------- GEMM, f16 A:  C[M,Nout] = A @ W + bias ----------------------
// grid.x = M/16, block = 32; NT = Nout/16; one wave owns a 16 x Nout row-block.
template <int NT>
__global__ void __launch_bounds__(32)
gemm_a16(const _Float16* __restrict__ A, const unsigned int* __restrict__ Bf,
         const float* __restrict__ bias, float* __restrict__ C,
         _Float16* __restrict__ C16, int M, int K, int Nout) {
  int lane = threadIdx.x & 31;
  int row0 = blockIdx.x * 16;
  v8f acc[NT];
#pragma unroll
  for (int nt = 0; nt < NT; ++nt) acc[nt] = (v8f){};
  int ksteps = K >> 5;
  for (int ks = 0; ks < ksteps; ++ks) {
    v16h a = load_a_f16(A, K, row0, ks, lane);
#pragma unroll
    for (int nt = 0; nt < NT; ++nt) {
      v16h b = load_b_frag(Bf, NT, ks, nt, lane);
      acc[nt] = wmma_f32_16x16x32(a, b, acc[nt]);
    }
  }
  int nlo = lane & 15;
  int mo  = (lane >> 4) * 8;
#pragma unroll
  for (int nt = 0; nt < NT; ++nt) {
    int n    = nt * 16 + nlo;
    float bb = bias ? bias[n] : 0.0f;
#pragma unroll
    for (int j = 0; j < 8; ++j) {
      float vv = acc[nt][j] + bb;
      size_t idx = (size_t)(row0 + mo + j) * Nout + n;
      C[idx] = vv;
      if (C16) C16[idx] = (_Float16)vv;
    }
  }
}

// ---------------- GEMM, f32 A (atomic-produced / raw input A) ----------------
template <int NT>
__global__ void __launch_bounds__(32)
gemm_a32(const float* __restrict__ A, const unsigned int* __restrict__ Bf,
         const float* __restrict__ bias, float* __restrict__ C,
         int M, int K, int Nout) {
  int lane = threadIdx.x & 31;
  int row0 = blockIdx.x * 16;
  v8f acc[NT];
#pragma unroll
  for (int nt = 0; nt < NT; ++nt) acc[nt] = (v8f){};
  int ksteps = K >> 5;
  for (int ks = 0; ks < ksteps; ++ks) {
    v16h a = load_a_f32(A, K, row0, ks, lane);
#pragma unroll
    for (int nt = 0; nt < NT; ++nt) {
      v16h b = load_b_frag(Bf, NT, ks, nt, lane);
      acc[nt] = wmma_f32_16x16x32(a, b, acc[nt]);
    }
  }
  int nlo = lane & 15;
  int mo  = (lane >> 4) * 8;
#pragma unroll
  for (int nt = 0; nt < NT; ++nt) {
    int n    = nt * 16 + nlo;
    float bb = bias ? bias[n] : 0.0f;
#pragma unroll
    for (int j = 0; j < 8; ++j)
      C[(size_t)(row0 + mo + j) * Nout + n] = acc[nt][j] + bb;
  }
}

// ---------------- fold We into Wq (Te trick) ---------------------------------
// Wqe[m,(hd,c)] = sum_d Wq[m,hd*32+d]*We[c,hd*32+d]; ce = same with bq.
__global__ void make_wqe(const float* __restrict__ Wq, const float* __restrict__ bq,
                         const float* __restrict__ We, float* __restrict__ Wqe,
                         float* __restrict__ ce) {
  int col = threadIdx.x;  // col = hd*32 + c
  int hd  = col >> 5;
  int c   = col & 31;
  for (int m = 0; m < HID; ++m) {
    float acc = 0.0f;
#pragma unroll 4
    for (int d = 0; d < HID; ++d)
      acc += Wq[m * HO + hd * HID + d] * We[c * HO + hd * HID + d];
    Wqe[m * HO + col] = acc;
  }
  float cb = 0.0f;
  for (int d = 0; d < HID; ++d)
    cb += bq[hd * HID + d] * We[c * HO + hd * HID + d];
  ce[col] = cb;
}

// ---------------- edge attention + scatter: one wave per edge ----------------
__global__ void __launch_bounds__(256)
edge_attn(const int* __restrict__ src, const int* __restrict__ dst,
          const float* __restrict__ ea, const float* __restrict__ q,
          const float* __restrict__ k, const float* __restrict__ v,
          const float* __restrict__ te, const float* __restrict__ be_l,
          float* __restrict__ agg, int E) {
  int wid  = (int)((blockIdx.x * (unsigned)blockDim.x + threadIdx.x) >> 5);
  int lane = threadIdx.x & 31;
  if (wid >= E) return;
  int i = dst[wid];
  int j = src[wid];
  float eav = ea[(size_t)wid * EDGE_DIM + lane];
  float s[HEADS];
#pragma unroll
  for (int hd = 0; hd < HEADS; ++hd) {
    int idx  = hd * HID + lane;
    float qv = q[(size_t)i * HO + idx];
    float kv = k[(size_t)j * HO + idx];
    float tv = te[(size_t)i * HO + idx];
    float bv = be_l[idx];
    s[hd] = qv * (kv + bv) + tv * eav;  // q.k + q.(ea@We) + q.be
  }
#pragma unroll
  for (int hd = 0; hd < HEADS; ++hd)
#pragma unroll
    for (int off = 16; off > 0; off >>= 1)
      s[hd] += __shfl_xor(s[hd], off, 32);

  const float scale = 0.17677669529663687f;  // 1/sqrt(HID)
  float a0 = s[0] * scale, a1 = s[1] * scale, a2 = s[2] * scale, a3 = s[3] * scale;
  float m  = fmaxf(fmaxf(a0, a1), fmaxf(a2, a3));
  float e0 = __expf(a0 - m), e1 = __expf(a1 - m);
  float e2 = __expf(a2 - m), e3 = __expf(a3 - m);
  float inv = 1.0f / (e0 + e1 + e2 + e3);
  float al[HEADS] = {e0 * inv, e1 * inv, e2 * inv, e3 * inv};
#pragma unroll
  for (int hd = 0; hd < HEADS; ++hd) {
    int idx = hd * HID + lane;
    atomicAdd(&agg[(size_t)i * HO + idx], v[(size_t)j * HO + idx] * al[hd]);
  }
}

// ---------------- residual + layernorm; also emits f16 copy of h -------------
__global__ void residual_ln(const float* __restrict__ pre, const float* __restrict__ g,
                            const float* __restrict__ b, float* __restrict__ h,
                            _Float16* __restrict__ h16, int N) {
  int i = blockIdx.x * blockDim.x + threadIdx.x;
  if (i >= N) return;
  float vv[HID];
  float mu = 0.0f;
#pragma unroll
  for (int d = 0; d < HID; ++d) {
    vv[d] = pre[(size_t)i * HID + d] + h[(size_t)i * HID + d];
    mu += vv[d];
  }
  mu *= (1.0f / HID);
  float var = 0.0f;
#pragma unroll
  for (int d = 0; d < HID; ++d) {
    float t = vv[d] - mu;
    var += t * t;
  }
  var *= (1.0f / HID);
  float inv = rsqrtf(var + 1e-5f);
#pragma unroll
  for (int d = 0; d < HID; ++d) {
    float o = (vv[d] - mu) * inv * g[d] + b[d];
    h[(size_t)i * HID + d]   = o;
    h16[(size_t)i * HID + d] = (_Float16)o;
  }
}

// ---------------- pooling ----------------------------------------------------
__device__ inline void atomicMaxF(float* addr, float val) {
  int* ai = (int*)addr;
  int old = __float_as_int(*addr);
  while (__int_as_float(old) < val) {
    int assumed = old;
    old = atomicCAS(ai, assumed, __float_as_int(val));
    if (old == assumed) break;
  }
}

__global__ void __launch_bounds__(32)
pool_accum(const float* __restrict__ h, const int* __restrict__ batch,
           float* __restrict__ means, float* __restrict__ maxs,
           float* __restrict__ gates, float* __restrict__ cnt) {
  int i = blockIdx.x;
  int d = threadIdx.x;
  int g = batch[i];
  float val = h[(size_t)i * HID + d];
  atomicAdd(&means[g * HID + d], val);
  atomicMaxF(&maxs[g * HID + d], val);
  atomicAdd(&gates[g * HID + d], val / (1.0f + __expf(-val)));  // x*sigmoid(x)
  if (d == 0) atomicAdd(&cnt[g], 1.0f);
}

__global__ void __launch_bounds__(128)
graph_mlp(const float* __restrict__ means, const float* __restrict__ maxs,
          const float* __restrict__ gates, const float* __restrict__ cnt,
          const float* __restrict__ W1, const float* __restrict__ b1,
          const float* __restrict__ W2, const float* __restrict__ b2,
          float* __restrict__ out) {
  __shared__ float repr[3 * HID];
  __shared__ float hid[HID];
  int g = blockIdx.x;
  int t = threadIdx.x;
  if (t < 3 * HID) {
    float c = fmaxf(cnt[g], 1.0f);
    float r;
    if (t < HID)          r = means[g * HID + t] / c;
    else if (t < 2 * HID) r = maxs[g * HID + (t - HID)];
    else                  r = gates[g * HID + (t - 2 * HID)] / c;
    repr[t] = r;
  }
  __syncthreads();
  if (t < HID) {
    float a = b1[t];
#pragma unroll 4
    for (int jj = 0; jj < 3 * HID; ++jj) a += repr[jj] * W1[jj * HID + t];
    hid[t] = fmaxf(a, 0.0f);
  }
  __syncthreads();
  float o = b2[t];
#pragma unroll 4
  for (int c = 0; c < HID; ++c) o += hid[c] * W2[c * OUT_DIM + t];
  out[g * OUT_DIM + t] = o;
}

// ---------------- utility ----------------------------------------------------
__global__ void fill_kernel(float* __restrict__ p, float val, long n) {
  long i = blockIdx.x * (long)blockDim.x + threadIdx.x;
  if (i < n) p[i] = val;
}
__global__ void copy_kernel(const float* __restrict__ s, float* __restrict__ d, long n) {
  long i = blockIdx.x * (long)blockDim.x + threadIdx.x;
  if (i < n) d[i] = s[i];
}

// ---------------- host orchestration -----------------------------------------
extern "C" void kernel_launch(void* const* d_in, const int* in_sizes, int n_in,
                              void* d_out, int out_size, void* d_ws, size_t ws_size,
                              hipStream_t stream) {
  (void)in_sizes; (void)n_in; (void)out_size; (void)ws_size;
  const float* x         = (const float*)d_in[0];
  const int*   eidx      = (const int*)  d_in[1];
  const float* edge_attr = (const float*)d_in[2];
  const int*   batch     = (const int*)  d_in[3];
  const float* W_in = (const float*)d_in[4];  const float* b_in = (const float*)d_in[5];
  const float* W_ep = (const float*)d_in[6];  const float* b_ep = (const float*)d_in[7];
  const float* Wq = (const float*)d_in[8];    const float* bq = (const float*)d_in[9];
  const float* Wk = (const float*)d_in[10];   const float* bk = (const float*)d_in[11];
  const float* Wv = (const float*)d_in[12];   const float* bv = (const float*)d_in[13];
  const float* We = (const float*)d_in[14];   const float* be = (const float*)d_in[15];
  const float* Wo = (const float*)d_in[16];   const float* bo = (const float*)d_in[17];
  const float* ln_g = (const float*)d_in[18]; const float* ln_b = (const float*)d_in[19];
  const float* W_o1 = (const float*)d_in[20]; const float* b_o1 = (const float*)d_in[21];
  const float* W_o2 = (const float*)d_in[22]; const float* b_o2 = (const float*)d_in[23];
  float* out = (float*)d_out;

  const int* src = eidx;            // edge_index[0]
  const int* dst = eidx + N_EDGES;  // edge_index[1]

  // ---------------- workspace layout (float units) ----------------
  float* ws = (float*)d_ws;
  float* h     = ws;                                   // N*HID
  float* tmp   = h    + (long)N_NODES * HID;           // N*HID
  float* ea    = tmp  + (long)N_NODES * HID;           // E*EDGE_DIM
  float* q     = ea   + (long)N_EDGES * EDGE_DIM;      // N*HO
  float* kbuf  = q    + (long)N_NODES * HO;
  float* vbuf  = kbuf + (long)N_NODES * HO;
  float* te    = vbuf + (long)N_NODES * HO;
  float* agg   = te   + (long)N_NODES * HO;
  float* wqe   = agg  + (long)N_NODES * HO;            // 32*128
  float* ce    = wqe  + HID * HO;                      // 128
  float* means = ce   + HO;                            // NG*HID
  float* maxs  = means + NG * HID;
  float* gates = maxs  + NG * HID;
  float* cnt   = gates + NG * HID;                     // NG
  _Float16* x16 = (_Float16*)(cnt + NG);               // N*IN_DIM halves
  _Float16* h16 = x16 + (long)N_NODES * IN_DIM;        // N*HID halves
  unsigned int* fW_in = (unsigned int*)(h16 + (long)N_NODES * HID);
  unsigned int* fW_ep = fW_in + (IN_DIM / 32) * (HID / 16) * 256;      // 4096
  unsigned int* fWq   = fW_ep + (EDGE_DIM / 32) * (EDGE_DIM / 16) * 256;
  unsigned int* fWk   = fWq + (HID / 32) * (HO / 16) * 256;            // 2048 each
  unsigned int* fWv   = fWk + (HID / 32) * (HO / 16) * 256;
  unsigned int* fWqe  = fWv + (HID / 32) * (HO / 16) * 256;
  unsigned int* fWo   = fWqe + (HID / 32) * (HO / 16) * 256;           // 4*2*256

  const int MT_N = N_NODES / 16;  // 3125
  const int MT_E = N_EDGES / 16;  // 50000

  // ---------------- prologue: convert x, pack static weights ----------------
  long nx = (long)N_NODES * IN_DIM;
  cvt_to_f16<<<(int)((nx + 255) / 256), 256, 0, stream>>>(x, x16, nx);
  pack_b_frags<<<(IN_DIM / 32) * (HID / 16), 32, 0, stream>>>(W_in, fW_in, IN_DIM, HID);
  pack_b_frags<<<(EDGE_DIM / 32) * (EDGE_DIM / 16), 32, 0, stream>>>(W_ep, fW_ep,
                                                                     EDGE_DIM, EDGE_DIM);

  // h = x @ W_in + b_in   (also emits h16)
  gemm_a16<HID / 16><<<MT_N, 32, 0, stream>>>(x16, fW_in, b_in, h, h16,
                                              N_NODES, IN_DIM, HID);
  // ea = edge_attr @ W_ep + b_ep  (f32 A path)
  gemm_a32<EDGE_DIM / 16><<<MT_E, 32, 0, stream>>>(edge_attr, fW_ep, b_ep, ea,
                                                   N_EDGES, EDGE_DIM, EDGE_DIM);

  for (int l = 0; l < L_LAYERS; ++l) {
    const float* Wq_l = Wq + (long)l * HID * HO;  const float* bq_l = bq + (long)l * HO;
    const float* Wk_l = Wk + (long)l * HID * HO;  const float* bk_l = bk + (long)l * HO;
    const float* Wv_l = Wv + (long)l * HID * HO;  const float* bv_l = bv + (long)l * HO;
    const float* We_l = We + (long)l * EDGE_DIM * HO; const float* be_l = be + (long)l * HO;
    const float* Wo_l = Wo + (long)l * HO * HID;  const float* bo_l = bo + (long)l * HID;

    make_wqe<<<1, HO, 0, stream>>>(Wq_l, bq_l, We_l, wqe, ce);

    const int qtiles = (HID / 32) * (HO / 16);  // 8
    pack_b_frags<<<qtiles, 32, 0, stream>>>(Wq_l, fWq, HID, HO);
    pack_b_frags<<<qtiles, 32, 0, stream>>>(Wk_l, fWk, HID, HO);
    pack_b_frags<<<qtiles, 32, 0, stream>>>(Wv_l, fWv, HID, HO);
    pack_b_frags<<<qtiles, 32, 0, stream>>>(wqe, fWqe, HID, HO);
    pack_b_frags<<<(HO / 32) * (HID / 16), 32, 0, stream>>>(Wo_l, fWo, HO, HID);

    gemm_a16<HO / 16><<<MT_N, 32, 0, stream>>>(h16, fWq, bq_l, q, nullptr,
                                               N_NODES, HID, HO);
    gemm_a16<HO / 16><<<MT_N, 32, 0, stream>>>(h16, fWk, bk_l, kbuf, nullptr,
                                               N_NODES, HID, HO);
    gemm_a16<HO / 16><<<MT_N, 32, 0, stream>>>(h16, fWv, bv_l, vbuf, nullptr,
                                               N_NODES, HID, HO);
    gemm_a16<HO / 16><<<MT_N, 32, 0, stream>>>(h16, fWqe, ce, te, nullptr,
                                               N_NODES, HID, HO);

    long nagg = (long)N_NODES * HO;
    fill_kernel<<<(int)((nagg + 255) / 256), 256, 0, stream>>>(agg, 0.0f, nagg);

    edge_attn<<<N_EDGES / 8, 256, 0, stream>>>(src, dst, ea, q, kbuf, vbuf, te,
                                               be_l, agg, N_EDGES);

    gemm_a32<HID / 16><<<MT_N, 32, 0, stream>>>(agg, fWo, bo_l, tmp,
                                                N_NODES, HO, HID);
    residual_ln<<<(N_NODES + 255) / 256, 256, 0, stream>>>(
        tmp, ln_g + (long)l * HID, ln_b + (long)l * HID, h, h16, N_NODES);
  }

  // ---------------- pooling + readout ----------------
  fill_kernel<<<(NG * HID + 255) / 256, 256, 0, stream>>>(means, 0.0f, NG * HID);
  fill_kernel<<<(NG * HID + 255) / 256, 256, 0, stream>>>(maxs, -3.0e38f, NG * HID);
  fill_kernel<<<(NG * HID + 255) / 256, 256, 0, stream>>>(gates, 0.0f, NG * HID);
  fill_kernel<<<1, 256, 0, stream>>>(cnt, 0.0f, NG);
  pool_accum<<<N_NODES, 32, 0, stream>>>(h, batch, means, maxs, gates, cnt);

  graph_mlp<<<NG, OUT_DIM, 0, stream>>>(means, maxs, gates, cnt,
                                        W_o1, b_o1, W_o2, b_o2, out);
  long nh = (long)N_NODES * HID;
  copy_kernel<<<(int)((nh + 255) / 256), 256, 0, stream>>>(h, out + NG * OUT_DIM, nh);
}
